// Encoder_68564857913539
// MI455X (gfx1250) — matrix operand order repaired and checked by reading, not compiled, not run on hardware
//
#include <hip/hip_runtime.h>
#include <hip/hip_bf16.h>

// ---------------------------------------------------------------------------
// CDNA5 (gfx1250) point-transformer encoder.
// All dense linear layers run through one WMMA f16->f32 GEMM kernel:
//   - 256-thread block = 8 waves, macro tile 64(M) x 128(N), K stepped by 64
//   - each wave computes a 32x32 sub-tile = 8 x v_wmma_f32_16x16x32_f16/iter
//   - weight tile staged via gfx1250 async copy (global_load_async_to_lds_b128
//     + s_wait_asynccnt), then converted f32->f16 transposed into LDS
//   - every fragment is a contiguous 16B LDS chunk (ds_load_b128)
// ---------------------------------------------------------------------------

typedef __attribute__((ext_vector_type(16))) _Float16 v16h;
typedef __attribute__((ext_vector_type(8)))  _Float16 v8h;
typedef __attribute__((ext_vector_type(4)))  _Float16 v4h;
typedef __attribute__((ext_vector_type(8)))  float    v8f;

#define HEADS 8
#define KNN   16

static __device__ __forceinline__ v16h cat8(v8h a, v8h b)
{
    v16h r;
#pragma unroll
    for (int i = 0; i < 8; ++i) { r[i] = a[i]; r[8 + i] = b[i]; }
    return r;
}

// ============================ WMMA GEMM ====================================
// C[M,N] = act(A[M,K] @ W[K,N] + bias)   (act: 0=none, 1=relu)
__global__ __launch_bounds__(256)
void k_gemm(const float* __restrict__ A, const float* __restrict__ W,
            const float* __restrict__ bias, float* __restrict__ C,
            int M, int K, int N, int act)
{
    __shared__ __align__(16) _Float16 As[64][72];    // 64 x 64 f16 (+pad)
    __shared__ __align__(16) _Float16 BsT[128][72];  // transposed: [n][k]
    __shared__ __align__(16) float    Wst[64][128];  // f32 weight staging (async)

    const int tid  = threadIdx.x;
    const int wave = tid >> 5;
    const int lane = tid & 31;
    const int wy   = wave >> 2;          // 0..1  -> M sub-tile (32 rows)
    const int wx   = wave & 3;           // 0..3  -> N sub-tile (32 cols)
    const int bm   = blockIdx.y * 64;
    const int bn   = blockIdx.x * 128;
    const int half = lane >> 4;
    const int l15  = lane & 15;

    v8f acc[2][2];
#pragma unroll
    for (int i = 0; i < 2; ++i)
#pragma unroll
        for (int j = 0; j < 2; ++j) {
            v8f z = {0.f, 0.f, 0.f, 0.f, 0.f, 0.f, 0.f, 0.f};
            acc[i][j] = z;
        }

    const bool kvec = (K & 63) == 0;     // all big GEMMs; false only for K=9 / K=3

    for (int kk = 0; kk < K; kk += 64) {
        // ---- stage A tile 64x64: branch-free, vectorized ----
        if (kvec) {
            // 64 rows x 16 float4 chunks = 1024, 4 per thread; no K guard
#pragma unroll
            for (int u = 0; u < 4; ++u) {
                int it = tid + u * 256;
                int r = it >> 4, cq = it & 15;
                int gr = bm + r;
                int grs = gr < M ? gr : M - 1;
                const float4 v = *(const float4*)&A[(size_t)grs * K + kk + cq * 4];
                bool ok = gr < M;
                v4h h;
                h[0] = (_Float16)(ok ? v.x : 0.f);
                h[1] = (_Float16)(ok ? v.y : 0.f);
                h[2] = (_Float16)(ok ? v.z : 0.f);
                h[3] = (_Float16)(ok ? v.w : 0.f);
                *(v4h*)&As[r][cq * 4] = h;
            }
        } else {
            // scalar clamped loads (K = 9 or 3), still branch-free
#pragma unroll
            for (int u = 0; u < 16; ++u) {
                int it = tid + u * 256;
                int r = it >> 6, c = it & 63;
                int gr = bm + r, gk = kk + c;
                int grs = gr < M ? gr : M - 1;
                int gks = gk < K ? gk : K - 1;
                float v = A[(size_t)grs * K + gks];
                As[r][c] = (_Float16)((gr < M && gk < K) ? v : 0.f);
            }
        }

        // ---- stage B tile 64(K) x 128(N) via gfx1250 async copy to LDS ----
        // 64*128 f32 = 2048 float4 chunks, 8 per thread; each thread then
        // converts/transposes only the chunks it loaded itself (no barrier
        // needed between wait and convert).
#pragma unroll
        for (int u = 0; u < 8; ++u) {
            int it = tid + u * 256;
            int r = it >> 5, nq = it & 31;      // r: k row (0..63), nq: float4 group
            int gk = kk + r;
            int gks = gk < K ? gk : K - 1;
            const float* gsrc = &W[(size_t)gks * N + bn + nq * 4];
            unsigned lds = (unsigned)(size_t)&Wst[r][nq * 4];  // low 32 bits = LDS offset
            asm volatile("global_load_async_to_lds_b128 %0, %1, off"
                         :: "v"(lds), "v"((unsigned long long)(size_t)gsrc)
                         : "memory");
        }
        asm volatile("s_wait_asynccnt 0" ::: "memory");
#pragma unroll
        for (int u = 0; u < 8; ++u) {
            int it = tid + u * 256;
            int r = it >> 5, nq = it & 31;
            bool ok = (kk + r) < K;
            const float4 v = *(const float4*)&Wst[r][nq * 4];
            int n0 = nq * 4;
            BsT[n0][r]     = (_Float16)(ok ? v.x : 0.f);
            BsT[n0 + 1][r] = (_Float16)(ok ? v.y : 0.f);
            BsT[n0 + 2][r] = (_Float16)(ok ? v.z : 0.f);
            BsT[n0 + 3][r] = (_Float16)(ok ? v.w : 0.f);
        }

        // gfx1250: prefetch next K tile (global_prefetch_b8)
        if (kk + 64 < K) {
            int pr = bm + (tid >> 2); pr = pr < M ? pr : (M - 1);
            __builtin_prefetch(&A[(size_t)pr * K + kk + 64], 0, 1);
            int pk = kk + 64 + (tid >> 3); pk = pk < K ? pk : (K - 1);
            __builtin_prefetch(&W[(size_t)pk * N + bn], 0, 1);
        }
        __syncthreads();

        // ---- fragments: contiguous 16B LDS chunks (ds_load_b128) ----
        const int ar0 = wy * 32 + l15;
        const int ar1 = ar0 + 16;
        const int bc0 = wx * 32 + l15;
        const int bc1 = bc0 + 16;
#pragma unroll
        for (int s = 0; s < 2; ++s) {
            const int ko = s * 32;
            v16h af0 = cat8(*(const v8h*)&As[ar0][ko + half * 8],
                            *(const v8h*)&As[ar0][ko + 16 + half * 8]);
            v16h af1 = cat8(*(const v8h*)&As[ar1][ko + half * 8],
                            *(const v8h*)&As[ar1][ko + 16 + half * 8]);
            v16h bf0 = cat8(*(const v8h*)&BsT[bc0][ko + half * 8],
                            *(const v8h*)&BsT[bc0][ko + 16 + half * 8]);
            v16h bf1 = cat8(*(const v8h*)&BsT[bc1][ko + half * 8],
                            *(const v8h*)&BsT[bc1][ko + 16 + half * 8]);

            acc[0][0] = __builtin_amdgcn_wmma_f32_16x16x32_f16(
                false, af0, false, bf0, (short)0, acc[0][0], false, false);
            acc[0][1] = __builtin_amdgcn_wmma_f32_16x16x32_f16(
                false, af0, false, bf1, (short)0, acc[0][1], false, false);
            acc[1][0] = __builtin_amdgcn_wmma_f32_16x16x32_f16(
                false, af1, false, bf0, (short)0, acc[1][0], false, false);
            acc[1][1] = __builtin_amdgcn_wmma_f32_16x16x32_f16(
                false, af1, false, bf1, (short)0, acc[1][1], false, false);
        }
        __syncthreads();
    }

    // ---- epilogue: lane holds col; VGPR r -> row r + 8*half ----
#pragma unroll
    for (int sm = 0; sm < 2; ++sm) {
#pragma unroll
        for (int sn = 0; sn < 2; ++sn) {
            int gn = bn + wx * 32 + sn * 16 + l15;
            if (gn >= N) continue;
            float bb = bias ? bias[gn] : 0.f;
#pragma unroll
            for (int r = 0; r < 8; ++r) {
                int gm = bm + wy * 32 + sm * 16 + r + half * 8;
                if (gm < M) {
                    float v = acc[sm][sn][r] + bb;
                    if (act == 1) v = fmaxf(v, 0.f);
                    C[(size_t)gm * N + gn] = v;
                }
            }
        }
    }
}

// ============================ LayerNorm (rows) =============================
__global__ __launch_bounds__(256)
void k_rowln(float* __restrict__ X, const float* __restrict__ s,
             const float* __restrict__ b, int d)
{
    __shared__ float red[256];
    const int row = blockIdx.x;
    const int t = threadIdx.x;
    float* x = X + (size_t)row * d;

    float sum = 0.f;
    for (int i = t; i < d; i += 256) sum += x[i];
    red[t] = sum; __syncthreads();
    for (int o = 128; o > 0; o >>= 1) { if (t < o) red[t] += red[t + o]; __syncthreads(); }
    const float mu = red[0] / (float)d;
    __syncthreads();

    float vs = 0.f;
    for (int i = t; i < d; i += 256) { float dd = x[i] - mu; vs += dd * dd; }
    red[t] = vs; __syncthreads();
    for (int o = 128; o > 0; o >>= 1) { if (t < o) red[t] += red[t + o]; __syncthreads(); }
    const float rstd = rsqrtf(red[0] / (float)d + 1e-5f);

    for (int i = t; i < d; i += 256)
        x[i] = (x[i] - mu) * rstd * s[i] + b[i];
}

// ============================ elementwise ==================================
__global__ void k_add_inplace(float* __restrict__ Y, const float* __restrict__ X, int n)
{
    int i = blockIdx.x * blockDim.x + threadIdx.x;
    if (i < n) Y[i] += X[i];
}

__global__ void k_i2f(const int* __restrict__ X, float* __restrict__ Y, int n)
{
    int i = blockIdx.x * blockDim.x + threadIdx.x;
    if (i < n) Y[i] = (float)X[i];
}

__global__ void k_gather_rows(const float* __restrict__ X, const int* __restrict__ idx,
                              float* __restrict__ Y, int m, int d)
{
    int t = blockIdx.x * blockDim.x + threadIdx.x;
    if (t >= m * d) return;
    int r = t / d, c = t - r * d;
    Y[t] = X[(size_t)idx[r] * d + c];
}

__global__ void k_gather_i32(const int* __restrict__ X, const int* __restrict__ idx,
                             int* __restrict__ Y, int m)
{
    int i = blockIdx.x * blockDim.x + threadIdx.x;
    if (i < m) Y[i] = X[idx[i]];
}

__global__ void k_bcast_add(float* __restrict__ X, const float* __restrict__ g, int m, int d)
{
    int t = blockIdx.x * blockDim.x + threadIdx.x;
    if (t >= m * d) return;
    X[t] += g[t % d];
}

// ============================ FPS (single block) ===========================
__global__ __launch_bounds__(1024)
void k_fps(const float* __restrict__ pos, int n, int nsamp,
           int* __restrict__ idx, float* __restrict__ dists)
{
    __shared__ float smax[1024];
    __shared__ int   simax[1024];
    __shared__ float lp[3];
    __shared__ int   cur;

    const int t = threadIdx.x;
    for (int i = t; i < n; i += 1024) dists[i] = 1e10f;
    if (t == 0) cur = 0;
    __syncthreads();

    for (int s = 0; s < nsamp; ++s) {
        if (t == 0) {
            int last = cur;
            idx[s] = last;
            lp[0] = pos[3 * last]; lp[1] = pos[3 * last + 1]; lp[2] = pos[3 * last + 2];
        }
        __syncthreads();
        float best = -1.f; int bi = 0;
        for (int i = t; i < n; i += 1024) {
            float dx = pos[3 * i] - lp[0];
            float dy = pos[3 * i + 1] - lp[1];
            float dz = pos[3 * i + 2] - lp[2];
            float d = dx * dx + dy * dy + dz * dz;
            float mn = fminf(dists[i], d);
            dists[i] = mn;
            if (mn > best) { best = mn; bi = i; }
        }
        smax[t] = best; simax[t] = bi;
        __syncthreads();
        for (int o = 512; o > 0; o >>= 1) {
            if (t < o && smax[t + o] > smax[t]) { smax[t] = smax[t + o]; simax[t] = simax[t + o]; }
            __syncthreads();
        }
        if (t == 0) cur = simax[0];
        __syncthreads();
    }
}

// ============================ kNN (thread per center) ======================
__global__ void k_knn(const float* __restrict__ pos, int m, int* __restrict__ src)
{
    int i = blockIdx.x * blockDim.x + threadIdx.x;
    if (i >= m) return;
    const float px = pos[3 * i], py = pos[3 * i + 1], pz = pos[3 * i + 2];
    float bd[KNN]; int bi[KNN];
#pragma unroll
    for (int e = 0; e < KNN; ++e) { bd[e] = 1e30f; bi[e] = 0; }
    for (int j = 0; j < m; ++j) {
        if (j == i) continue;
        float dx = pos[3 * j] - px, dy = pos[3 * j + 1] - py, dz = pos[3 * j + 2] - pz;
        float d = dx * dx + dy * dy + dz * dz;
        if (d < bd[KNN - 1]) {
            int e = KNN - 1;
            while (e > 0 && bd[e - 1] > d) { bd[e] = bd[e - 1]; bi[e] = bi[e - 1]; --e; }
            bd[e] = d; bi[e] = j;
        }
    }
#pragma unroll
    for (int e = 0; e < KNN; ++e) src[i * KNN + e] = bi[e];
}

// ============================ edge pos deltas ==============================
__global__ void k_edge_pd(const float* __restrict__ cp, const int* __restrict__ src,
                          float* __restrict__ pd, int E)
{
    int e = blockIdx.x * blockDim.x + threadIdx.x;
    if (e >= E) return;
    int i = e / KNN;          // dst (center)
    int j = src[e];           // src (neighbor)
    pd[3 * e]     = cp[3 * i]     - cp[3 * j];
    pd[3 * e + 1] = cp[3 * i + 1] - cp[3 * j + 1];
    pd[3 * e + 2] = cp[3 * i + 2] - cp[3 * j + 2];
}

// ======== attention: per (center, head) register softmax over K=16 =========
__global__ void k_attn(const float* __restrict__ qn, const float* __restrict__ kn,
                       const float* __restrict__ vn, const float* __restrict__ pe,
                       const int* __restrict__ src, float* __restrict__ aggr,
                       int m, int cout, int hd)
{
    int t = blockIdx.x * blockDim.x + threadIdx.x;
    if (t >= m * HEADS) return;
    const int i = t / HEADS, h = t - i * HEADS;
    const float scale = rsqrtf((float)hd);
    const size_t qoff = (size_t)i * cout + h * hd;

    float sc[KNN];
    float mx = -1e30f;
#pragma unroll
    for (int e = 0; e < KNN; ++e) {
        const int eid = i * KNN + e;
        const int j = src[eid];
        const size_t koff = (size_t)j * cout + h * hd;
        const size_t poff = (size_t)eid * cout + h * hd;
        float dot = 0.f;
        for (int d = 0; d < hd; ++d)
            dot += qn[qoff + d] * (kn[koff + d] + pe[poff + d]);
        sc[e] = dot * scale;
        mx = fmaxf(mx, sc[e]);
    }
    float ssum = 0.f;
#pragma unroll
    for (int e = 0; e < KNN; ++e) { sc[e] = expf(sc[e] - mx); ssum += sc[e]; }
    const float inv = 1.f / (ssum + 1e-16f);

    for (int d = 0; d < hd; ++d) {
        float a = 0.f;
#pragma unroll
        for (int e = 0; e < KNN; ++e)
            a += sc[e] * vn[(size_t)src[i * KNN + e] * cout + h * hd + d];
        aggr[qoff + d] = a * inv;
    }
}

// ============================ virtual node helpers =========================
__global__ __launch_bounds__(256)
void k_colmean(const float* __restrict__ X, float* __restrict__ out, int m, int d)
{
    __shared__ float red[256];
    const int col = blockIdx.x;
    const int t = threadIdx.x;
    float s = 0.f;
    for (int r = t; r < m; r += 256) s += X[(size_t)r * d + col];
    red[t] = s; __syncthreads();
    for (int o = 128; o > 0; o >>= 1) { if (t < o) red[t] += red[t + o]; __syncthreads(); }
    if (t == 0) out[col] = red[0] / (float)m;
}

__global__ void k_veclin(const float* __restrict__ x, const float* __restrict__ W,
                         const float* __restrict__ b, float* __restrict__ y,
                         int din, int dout)
{
    int o = blockIdx.x * blockDim.x + threadIdx.x;
    if (o >= dout) return;
    float s = b[o];
    for (int i = 0; i < din; ++i) s += x[i] * W[(size_t)i * dout + o];
    y[o] = s;
}

// ============================ host side ====================================
struct Lin { const float *w, *b; };
struct LNp { const float *s, *b; };
struct BlockP {
    Lin ffn1, ffn2, k, out, pe1, pe2, q, res, v;
    LNp ln1, ln2, pe_ln;
    bool has_res;
};
struct VNp { Lin agg, dist; LNp ln; };

struct Arena {
    char* base; size_t off;
    float* f(size_t n) { float* p = (float*)(base + off); off += ((n * 4 + 255) / 256) * 256; return p; }
    int*   i(size_t n) { int*   p = (int*)(base + off);  off += ((n * 4 + 255) / 256) * 256; return p; }
};

static void gemm(hipStream_t st, const float* A, const float* W, const float* b,
                 float* C, int M, int K, int N, int act)
{
    dim3 g((N + 127) / 128, (M + 63) / 64);
    k_gemm<<<g, 256, 0, st>>>(A, W, b, C, M, K, N, act);
}

static void rowln(hipStream_t st, float* X, const LNp& p, int m, int d)
{
    k_rowln<<<m, 256, 0, st>>>(X, p.s, p.b, d);
}

static void vnode(hipStream_t st, const VNp& p, float* X, int m, int d, Arena a)
{
    float* g0 = a.f(d); float* g1 = a.f(d); float* g2 = a.f(d);
    k_colmean<<<d, 256, 0, st>>>(X, g0, m, d);
    k_veclin<<<(d + 255) / 256, 256, 0, st>>>(g0, p.agg.w, p.agg.b, g1, d, d);
    k_rowln<<<1, 256, 0, st>>>(g1, p.ln.s, p.ln.b, d);
    k_veclin<<<(d + 255) / 256, 256, 0, st>>>(g1, p.dist.w, p.dist.b, g2, d, d);
    k_bcast_add<<<(m * d + 255) / 256, 256, 0, st>>>(X, g2, m, d);
}

// One attention-graph-transformer block: cx[m,cin] -> outx[m,cout]
static void agt_block(hipStream_t st, const BlockP& p, const float* cx,
                      const float* cp, const int* src, float* outx,
                      int m, int cin, int cout, Arena a)
{
    const int E = m * KNN;
    const float* resp;
    if (p.has_res) {
        float* rb = a.f((size_t)m * cout);
        gemm(st, cx, p.res.w, p.res.b, rb, m, cin, cout, 0);
        resp = rb;
    } else {
        resp = cx;   // cin == cout
    }
    float* qn = a.f((size_t)m * cout);
    float* kn = a.f((size_t)m * cout);
    float* vn = a.f((size_t)m * cout);
    gemm(st, cx, p.q.w, p.q.b, qn, m, cin, cout, 0);
    gemm(st, cx, p.k.w, p.k.b, kn, m, cin, cout, 0);
    gemm(st, cx, p.v.w, p.v.b, vn, m, cin, cout, 0);

    float* pd  = a.f((size_t)E * 3);
    float* peh = a.f((size_t)E * cout);
    float* pe  = a.f((size_t)E * cout);
    k_edge_pd<<<(E + 255) / 256, 256, 0, st>>>(cp, src, pd, E);
    gemm(st, pd, p.pe1.w, p.pe1.b, peh, E, 3, cout, 1);        // relu
    gemm(st, peh, p.pe2.w, p.pe2.b, pe, E, cout, cout, 0);
    rowln(st, pe, p.pe_ln, E, cout);

    float* aggr = a.f((size_t)m * cout);
    k_attn<<<(m * HEADS + 255) / 256, 256, 0, st>>>(qn, kn, vn, pe, src, aggr,
                                                    m, cout, cout / HEADS);

    float* hb = a.f((size_t)m * cout);
    gemm(st, aggr, p.out.w, p.out.b, hb, m, cout, cout, 0);
    k_add_inplace<<<(m * cout + 255) / 256, 256, 0, st>>>(hb, resp, m * cout);
    rowln(st, hb, p.ln1, m, cout);

    float* fh = a.f((size_t)m * 4 * cout);
    gemm(st, hb, p.ffn1.w, p.ffn1.b, fh, m, cout, 4 * cout, 1);  // relu
    gemm(st, fh, p.ffn2.w, p.ffn2.b, outx, m, 4 * cout, cout, 0);
    k_add_inplace<<<(m * cout + 255) / 256, 256, 0, st>>>(outx, hb, m * cout);
    rowln(st, outx, p.ln2, m, cout);
}

extern "C" void kernel_launch(void* const* d_in, const int* in_sizes, int n_in,
                              void* d_out, int out_size, void* d_ws, size_t ws_size,
                              hipStream_t stream)
{
    (void)n_in; (void)out_size; (void)ws_size;
    const int IN_DIM = 9;
    const int N  = in_sizes[0] / IN_DIM;   // 32768
    const int S1 = 8192, S2 = 2048;
    const int D0 = 128, D1 = 256, D2 = 512;

    const float* x      = (const float*)d_in[0];
    const float* pos    = (const float*)d_in[1];
    const int*   labels = (const int*)d_in[2];
    const int*   batch  = (const int*)d_in[3];

    // ---- unpack params in JAX pytree flatten order (sorted dict keys) ----
    int ci = 4;
    auto nf   = [&]() -> const float* { return (const float*)d_in[ci++]; };
    auto rLin = [&]() { Lin l; l.b = nf(); l.w = nf(); return l; };          // b < w
    auto rLN  = [&]() { LNp l; l.b = nf(); l.s = nf(); return l; };          // b < s
    auto rBlk = [&](bool has_res) {
        BlockP p{};
        p.ffn1 = rLin(); p.ffn2 = rLin(); p.k = rLin();
        p.ln1 = rLN(); p.ln2 = rLN(); p.out = rLin();
        p.pe1 = rLin(); p.pe2 = rLin(); p.pe_ln = rLN();
        p.q = rLin();
        if (has_res) p.res = rLin();
        p.v = rLin();
        p.has_res = has_res;
        return p;
    };
    auto rVN = [&]() { VNp v; v.agg = rLin(); v.dist = rLin(); v.ln = rLN(); return v; };

    // top-level params keys sorted: stage0, stage1, stage2, vn
    Lin s0lin = rLin(); LNp s0ln = rLN();                 // stage0: lin, ln
    BlockP b10 = rBlk(true);  BlockP b11 = rBlk(false);   // stage1
    BlockP b20 = rBlk(true);  BlockP b21 = rBlk(false);   // stage2
    VNp vn0 = rVN(); VNp vn1 = rVN(); VNp vn2 = rVN();    // vn list

    // ---- workspace ----
    Arena ar{(char*)d_ws, 0};
    float* h0    = ar.f((size_t)N * D0);        // stage0 features (feats[1])
    float* fA    = ar.f((size_t)S1 * D2);       // feature ping
    float* fB    = ar.f((size_t)S1 * D2);       // feature pong
    float* cp1   = ar.f((size_t)S1 * 3);
    float* cp2   = ar.f((size_t)S2 * 3);
    int*   cl1   = ar.i(S1); int* cb1 = ar.i(S1);
    int*   cl2   = ar.i(S2); int* cb2 = ar.i(S2);
    int*   fidx  = ar.i(S1);
    float* fdist = ar.f(N);
    int*   src   = ar.i((size_t)S1 * KNN);
    Arena scratch = ar;                          // per-op scratch starts here

    // ---- stage 0: MLP 9->128 + LN + vnode ----
    gemm(stream, x, s0lin.w, s0lin.b, h0, N, IN_DIM, D0, 1);   // relu
    rowln(stream, h0, s0ln, N, D0);
    vnode(stream, vn0, h0, N, D0, scratch);

    // ---- stage 1: FPS(8192) -> kNN -> 2 blocks -> vnode ----
    k_fps<<<1, 1024, 0, stream>>>(pos, N, S1, fidx, fdist);
    k_gather_rows<<<(S1 * D0 + 255) / 256, 256, 0, stream>>>(h0, fidx, fA, S1, D0);
    k_gather_rows<<<(S1 * 3 + 255) / 256, 256, 0, stream>>>(pos, fidx, cp1, S1, 3);
    k_gather_i32<<<(S1 + 255) / 256, 256, 0, stream>>>(labels, fidx, cl1, S1);
    k_gather_i32<<<(S1 + 255) / 256, 256, 0, stream>>>(batch, fidx, cb1, S1);
    k_knn<<<(S1 + 127) / 128, 128, 0, stream>>>(cp1, S1, src);
    agt_block(stream, b10, fA, cp1, src, fB, S1, D0, D1, scratch);
    agt_block(stream, b11, fB, cp1, src, fA, S1, D1, D1, scratch);
    vnode(stream, vn1, fA, S1, D1, scratch);
    float* f1 = fA;   // stage1 features [S1, 256]

    // ---- stage 2: FPS(2048) -> kNN -> 2 blocks -> vnode ----
    k_fps<<<1, 1024, 0, stream>>>(cp1, S1, S2, fidx, fdist);
    k_gather_rows<<<(S2 * D1 + 255) / 256, 256, 0, stream>>>(f1, fidx, fB, S2, D1);
    k_gather_rows<<<(S2 * 3 + 255) / 256, 256, 0, stream>>>(cp1, fidx, cp2, S2, 3);
    k_gather_i32<<<(S2 + 255) / 256, 256, 0, stream>>>(cl1, fidx, cl2, S2);
    k_gather_i32<<<(S2 + 255) / 256, 256, 0, stream>>>(cb1, fidx, cb2, S2);
    k_knn<<<(S2 + 127) / 128, 128, 0, stream>>>(cp2, S2, src);
    // fB holds stage2 input; need a second stage-2 buffer distinct from f1
    float* g2buf = scratch.f((size_t)S2 * D2);
    Arena scratch2 = scratch;
    agt_block(stream, b20, fB, cp2, src, g2buf, S2, D1, D2, scratch2);
    agt_block(stream, b21, g2buf, cp2, src, fB, S2, D2, D2, scratch2);
    vnode(stream, vn2, fB, S2, D2, scratch2);
    float* f2 = fB;   // stage2 features [S2, 512]

    // ---- assemble outputs (floats; int outputs converted) ----
    float* out = (float*)d_out;
    size_t o = 0;
    hipMemcpyAsync(out + o, x,  (size_t)N * 9 * 4,   hipMemcpyDeviceToDevice, stream); o += (size_t)N * 9;
    hipMemcpyAsync(out + o, h0, (size_t)N * D0 * 4,  hipMemcpyDeviceToDevice, stream); o += (size_t)N * D0;
    hipMemcpyAsync(out + o, f1, (size_t)S1 * D1 * 4, hipMemcpyDeviceToDevice, stream); o += (size_t)S1 * D1;
    hipMemcpyAsync(out + o, f2, (size_t)S2 * D2 * 4, hipMemcpyDeviceToDevice, stream); o += (size_t)S2 * D2;
    hipMemcpyAsync(out + o, pos, (size_t)N * 3 * 4,  hipMemcpyDeviceToDevice, stream); o += (size_t)N * 3;
    hipMemcpyAsync(out + o, pos, (size_t)N * 3 * 4,  hipMemcpyDeviceToDevice, stream); o += (size_t)N * 3;
    hipMemcpyAsync(out + o, cp1, (size_t)S1 * 3 * 4, hipMemcpyDeviceToDevice, stream); o += (size_t)S1 * 3;
    hipMemcpyAsync(out + o, cp2, (size_t)S2 * 3 * 4, hipMemcpyDeviceToDevice, stream); o += (size_t)S2 * 3;
    k_i2f<<<(N + 255) / 256, 256, 0, stream>>>(labels, out + o, N); o += N;
    k_i2f<<<(N + 255) / 256, 256, 0, stream>>>(labels, out + o, N); o += N;
    k_i2f<<<(S1 + 255) / 256, 256, 0, stream>>>(cl1, out + o, S1);  o += S1;
    k_i2f<<<(S2 + 255) / 256, 256, 0, stream>>>(cl2, out + o, S2);  o += S2;
    k_i2f<<<(N + 255) / 256, 256, 0, stream>>>(batch, out + o, N);  o += N;
    k_i2f<<<(N + 255) / 256, 256, 0, stream>>>(batch, out + o, N);  o += N;
    k_i2f<<<(S1 + 255) / 256, 256, 0, stream>>>(cb1, out + o, S1);  o += S1;
    k_i2f<<<(S2 + 255) / 256, 256, 0, stream>>>(cb2, out + o, S2);  o += S2;
}
